// GCMKGATCL_Ablation_35553739276538
// MI455X (gfx1250) — matrix operand order repaired and checked by reading, not compiled, not run on hardware
//
#include <hip/hip_runtime.h>

// ---------------------------------------------------------------------------
// Types for CDNA5 WMMA (wave32, v_wmma_f32_16x16x32_bf16)
// ---------------------------------------------------------------------------
typedef __bf16 bf16_t;
typedef __attribute__((ext_vector_type(16))) __bf16 v16bf;
typedef __attribute__((ext_vector_type(8)))  __bf16 v8bf;
typedef __attribute__((ext_vector_type(8)))  float  v8f;

__device__ __forceinline__ v8f wmma_bf16(v16bf a, v16bf b, v8f c) {
    // 8 args: (neg_a, A, neg_b, B, c_mod, C, reuse_a, reuse_b)
    return __builtin_amdgcn_wmma_f32_16x16x32_bf16(
        false, a, false, b, (short)0, c, false, false);
}

// A-operand gather, contiguous k order (k = 16*half + s): p points at row start + kk
__device__ __forceinline__ v16bf load_a(const bf16_t* __restrict__ p, int half) {
    const v8bf lo = *(const v8bf*)(p + half * 16);
    const v8bf hi = *(const v8bf*)(p + half * 16 + 8);
    v16bf a;
#pragma unroll
    for (int j = 0; j < 8; ++j) { a[j] = lo[j]; a[8 + j] = hi[j]; }
    return a;
}

// Model dims
#define NUSERS 2500
#define NITEMS 3000
#define NENT   2500
#define NTOT   8000
#define DIM    128
#define NEDGE  256000
#define TOPK   16

enum {
    F_BIAS   = 1,   // add bias[col]
    F_RELU   = 2,   // relu epilogue
    F_ACC    = 4,   // C += result (read f32 C)
    F_DOUBLE = 8,   // result *= 2 (folded cross-modal attention)
    F_OF32   = 16,  // write f32 C
    F_OB16   = 32,  // write bf16 Cb
};

// ---------------------------------------------------------------------------
// WMMA GEMM:  C[M,128] (op)= act( A[M,K](bf16) @ W[K,128] + b )
// K/LDW compile-time; K-loop strip-mined in 128-wide macro-chunks: batch all
// loads, then a sched_barrier pins them before the 4-WMMA chain so tail loads
// stay in flight under matrix math instead of each WMMA waiting loadcnt==0.
// Wt is the weight transposed+converted to [128][LDW] bf16 so each B K-chunk
// is one contiguous 32B vector load. One wave -> one 16x16 tile, f32 accum.
// ---------------------------------------------------------------------------
template <int FLAGS, int K, int LDW>
__global__ __launch_bounds__(32)
void k_gemm128(const bf16_t* __restrict__ A,   // [M,K] bf16 row-major
               const bf16_t* __restrict__ Wt,  // [128][LDW] bf16 (W transposed)
               const float*  __restrict__ bias,
               float*  __restrict__ C,         // [M,128] f32
               bf16_t* __restrict__ Cb,        // [M,128] bf16
               int M)
{
    static_assert(K % 128 == 0, "K must be a multiple of 128");
    const int lane = threadIdx.x;
    const int half = lane >> 4;
    const int mloc = lane & 15;
    const int rowbase = blockIdx.x * 16;
    const int col = blockIdx.y * 16 + mloc;

    int arow = rowbase + mloc;
    if (arow >= M) arow = M - 1;   // clamp for partial tiles (stores guarded)

    const bf16_t* pa = A + (size_t)arow * K;
    const bf16_t* pw = Wt + (size_t)col * LDW + half * 16;

    v8f acc = {0.f, 0.f, 0.f, 0.f, 0.f, 0.f, 0.f, 0.f};
#pragma unroll
    for (int kb = 0; kb < K; kb += 128) {
        v16bf av[4], bv[4];
#pragma unroll
        for (int t = 0; t < 4; ++t) {
            av[t] = load_a(pa + kb + t * 32, half);
            bv[t] = *(const v16bf*)(pw + kb + t * 32);
        }
        __builtin_amdgcn_sched_barrier(0);   // keep all loads issued first
#pragma unroll
        for (int t = 0; t < 4; ++t) acc = wmma_bf16(av[t], bv[t], acc);
    }

    float bv = 0.f;
    if constexpr (FLAGS & F_BIAS) bv = bias[col];
#pragma unroll
    for (int r = 0; r < 8; ++r) {
        const int row = rowbase + half * 8 + r;
        if (row >= M) continue;
        float v = acc[r] + bv;
        if constexpr (FLAGS & F_DOUBLE) v *= 2.f;
        if constexpr (FLAGS & F_ACC)    v += C[(size_t)row * DIM + col];
        if constexpr (FLAGS & F_RELU)   v = fmaxf(v, 0.f);
        if constexpr (FLAGS & F_OF32)   C[(size_t)row * DIM + col] = v;
        if constexpr (FLAGS & F_OB16)   Cb[(size_t)row * DIM + col] = (__bf16)v;
    }
}

// ---------------------------------------------------------------------------
// Fused scores + top-16:  scores = (qn @ kn^T)/sqrt(128); per-row top-16
// -> softmax weights + indices. One wave owns 16 rows; 32 columns per
// iteration: 16 b128 loads pinned (sched_barrier) ahead of the 8-WMMA chain,
// plus a global_prefetch of the kn rows two iterations ahead.
// Running-min-cached top-16 lists live in LDS; the 256MB/layer scores matrix
// never touches global memory.
// ---------------------------------------------------------------------------
__global__ __launch_bounds__(32)
void k_scores_topk(const bf16_t* __restrict__ qn,   // [NTOT,128] bf16
                   const bf16_t* __restrict__ kn,   // [NTOT,128] bf16
                   float* __restrict__ topw,        // [NTOT,16]
                   int*   __restrict__ topi,        // [NTOT,16]
                   int N)
{
    __shared__ float tile [16][33];
    __shared__ float svals[16][17];
    __shared__ int   sidx [16][17];

    const int lane = threadIdx.x;
    const int half = lane >> 4;
    const int mloc = lane & 15;
    const int rowbase = blockIdx.x * 16;

    // Preload A (16 rows x K=128) once: 4 K-chunks of 32
    const bf16_t* qrow = qn + (size_t)(rowbase + mloc) * DIM;
    v16bf a[4];
#pragma unroll
    for (int t = 0; t < 4; ++t) a[t] = load_a(qrow + t * 32, half);

    if (lane < 16) {
#pragma unroll
        for (int s = 0; s < TOPK; ++s) { svals[lane][s] = -3.0e38f; sidx[lane][s] = 0; }
    }
    float curmin = -3.0e38f;
    int   curminpos = 0;
    __syncthreads();

    const float scale = 0.08838834764831845f;  // 1/sqrt(128)

    for (int c = 0; c < N; c += 32) {
        // speculative prefetch of the column rows two iterations ahead:
        // 32 lanes cover exactly the 32 rows of iteration c+64
        int cpf = c + 64;
        if (cpf > N - 32) cpf = N - 32;
        __builtin_prefetch(kn + (size_t)(cpf + lane) * DIM, 0, 1);

        // batch-load both 16-wide column tiles (16 x b128) before any WMMA
        const bf16_t* krow0 = kn + (size_t)(c + mloc) * DIM + half * 16;
        const bf16_t* krow1 = kn + (size_t)(c + 16 + mloc) * DIM + half * 16;
        v16bf b[8];
#pragma unroll
        for (int t = 0; t < 4; ++t) {
            b[t]     = *(const v16bf*)(krow0 + t * 32);
            b[4 + t] = *(const v16bf*)(krow1 + t * 32);
        }
        __builtin_amdgcn_sched_barrier(0);   // pin loads ahead of WMMA chain

        v8f cc0 = {0.f, 0.f, 0.f, 0.f, 0.f, 0.f, 0.f, 0.f};
        v8f cc1 = {0.f, 0.f, 0.f, 0.f, 0.f, 0.f, 0.f, 0.f};
#pragma unroll
        for (int t = 0; t < 4; ++t) cc0 = wmma_bf16(a[t], b[t], cc0);
#pragma unroll
        for (int t = 0; t < 4; ++t) cc1 = wmma_bf16(a[t], b[4 + t], cc1);

#pragma unroll
        for (int r = 0; r < 8; ++r) {
            tile[half * 8 + r][mloc]      = cc0[r] * scale;
            tile[half * 8 + r][16 + mloc] = cc1[r] * scale;
        }
        __syncthreads();

        if (lane < 16) {
            for (int j = 0; j < 32; ++j) {
                const float s = tile[lane][j];
                if (s > curmin) {
                    svals[lane][curminpos] = s;
                    sidx [lane][curminpos] = c + j;
                    float mn = svals[lane][0]; int mp = 0;
#pragma unroll
                    for (int t = 1; t < TOPK; ++t) {
                        const float v = svals[lane][t];
                        if (v < mn) { mn = v; mp = t; }
                    }
                    curmin = mn; curminpos = mp;
                }
            }
        }
        __syncthreads();
    }

    // softmax over the (unordered) top-16 set — order-invariant downstream
    if (lane < 16) {
        const int row = rowbase + lane;
        float mx = -3.0e38f;
#pragma unroll
        for (int s = 0; s < TOPK; ++s) mx = fmaxf(mx, svals[lane][s]);
        float ex[TOPK];
        float se = 0.f;
#pragma unroll
        for (int s = 0; s < TOPK; ++s) { ex[s] = __expf(svals[lane][s] - mx); se += ex[s]; }
        const float inv = 1.f / se;
#pragma unroll
        for (int s = 0; s < TOPK; ++s) {
            topw[(size_t)row * TOPK + s] = ex[s] * inv;
            topi[(size_t)row * TOPK + s] = sidx[lane][s];
        }
    }
}

// ---------------------------------------------------------------------------
// Sparse segment-sum:  side[row[e], :] += data[e] * emb[col[e], :]
// ---------------------------------------------------------------------------
__global__ __launch_bounds__(128)
void k_segsum(const float* __restrict__ data, const int* __restrict__ rows,
              const int* __restrict__ cols, const float* __restrict__ emb,
              float* __restrict__ side)
{
    const int e = blockIdx.x;
    const int d = threadIdx.x;
    const float v = data[e] * emb[(size_t)cols[e] * DIM + d];
    unsafeAtomicAdd(&side[(size_t)rows[e] * DIM + d], v);
}

// out[row,:] = relu( sum_k w[row,k] * vn[ti[row,k], :] )   (f32 + bf16 copies)
__global__ __launch_bounds__(128)
void k_agg_relu(const float* __restrict__ vn, const float* __restrict__ topw,
                const int* __restrict__ topi, float* __restrict__ out,
                bf16_t* __restrict__ outb)
{
    const int row = blockIdx.x;
    const int d = threadIdx.x;
    float acc = 0.f;
#pragma unroll
    for (int k = 0; k < TOPK; ++k) {
        const float w = topw[(size_t)row * TOPK + k];
        const int idx = topi[(size_t)row * TOPK + k];
        acc += w * vn[(size_t)idx * DIM + d];
    }
    const float v = fmaxf(acc, 0.f);
    out [(size_t)row * DIM + d] = v;
    outb[(size_t)row * DIM + d] = (__bf16)v;
}

// Gated fusion: out = mg*id + (1-mg)*mm + 0.1*id; mg = sig(sig(g)/0.5)
__global__ void k_fuse(const float* __restrict__ idemb, const float* __restrict__ mm,
                       const float* __restrict__ glin, float* __restrict__ out,
                       bf16_t* __restrict__ outb, int n)
{
    const int i = blockIdx.x * blockDim.x + threadIdx.x;
    if (i >= n) return;
    const float g  = glin[i];
    const float s1 = 1.f / (1.f + __expf(-g));
    const float mg = 1.f / (1.f + __expf(-2.f * s1));
    const float iv = idemb[i], mv = mm[i];
    const float v = mg * iv + (1.f - mg) * mv + 0.1f * iv;
    out[i]  = v;
    outb[i] = (__bf16)v;
}

__global__ void k_relmean(const float* __restrict__ rel, float* __restrict__ out)
{
    const int d = threadIdx.x;
    float s = 0.f;
#pragma unroll
    for (int r = 0; r < 32; ++r) s += rel[r * DIM + d];
    out[d] = s * (1.f / 32.f);
}

// sidesc_bf = bf16( side * rel_mean[col] )
__global__ void k_scalecols_bf(const float* __restrict__ in, const float* __restrict__ sc,
                               bf16_t* __restrict__ out, int n)
{
    const int i = blockIdx.x * blockDim.x + threadIdx.x;
    if (i < n) out[i] = (__bf16)(in[i] * sc[i & (DIM - 1)]);
}

__global__ void k_tobf16(const float* __restrict__ in, bf16_t* __restrict__ out, int n)
{
    const int i = blockIdx.x * blockDim.x + threadIdx.x;
    if (i < n) out[i] = (__bf16)in[i];
}

// Weight transpose + convert: Wt[n][k] = bf16(W[k][n]);  W is [K,128]
__global__ void k_wtrans(const float* __restrict__ W, bf16_t* __restrict__ Wt, int K)
{
    const int i = blockIdx.x * blockDim.x + threadIdx.x;
    if (i < K * DIM) {
        const int k = i / DIM, n = i % DIM;
        Wt[(size_t)n * K + k] = (__bf16)W[i];
    }
}

// Row L2-normalize: out[row,:] = in[row,:] / max(||in[row,:]||, 1e-12)
__global__ __launch_bounds__(128)
void k_norm(const float* __restrict__ in, float* __restrict__ out)
{
    __shared__ float red[128];
    const int row = blockIdx.x;
    const int d = threadIdx.x;
    const float v = in[(size_t)row * DIM + d];
    red[d] = v * v;
    __syncthreads();
    for (int s = 64; s > 0; s >>= 1) {
        if (d < s) red[d] += red[d + s];
        __syncthreads();
    }
    const float den = fmaxf(sqrtf(red[0]), 1e-12f);
    out[(size_t)row * DIM + d] = v / den;
}

// ---------------------------------------------------------------------------
// Orchestration
// ---------------------------------------------------------------------------
extern "C" void kernel_launch(void* const* d_in, const int* in_sizes, int n_in,
                              void* d_out, int out_size, void* d_ws, size_t ws_size,
                              hipStream_t stream)
{
    const float* user_w = (const float*)d_in[0];
    const float* item_w = (const float*)d_in[1];
    const float* ent_w  = (const float*)d_in[2];
    const float* rel_w  = (const float*)d_in[3];
    const float* mfeat  = (const float*)d_in[4];
    const float* ufeat  = (const float*)d_in[5];
    const float* me_w1  = (const float*)d_in[6];
    const float* me_b1  = (const float*)d_in[7];
    const float* me_w2  = (const float*)d_in[8];
    const float* me_b2  = (const float*)d_in[9];
    const float* ue_w1  = (const float*)d_in[10];
    const float* ue_b1  = (const float*)d_in[11];
    const float* ue_w2  = (const float*)d_in[12];
    const float* ue_b2  = (const float*)d_in[13];
    // cm_qw/cm_qb/cm_kw/cm_kb (14..17) cancel exactly: identical modalities
    // make the 2x2 attention softmax uniform -> ao == v.
    const float* cm_vw  = (const float*)d_in[18];
    const float* cm_vb  = (const float*)d_in[19];
    const float* cm_ow  = (const float*)d_in[20];
    const float* cm_ob  = (const float*)d_in[21];
    const float* mg_w   = (const float*)d_in[22];
    const float* mg_b   = (const float*)d_in[23];
    const float* ug_w   = (const float*)d_in[24];
    const float* ug_b   = (const float*)d_in[25];
    const float* gnn_qw = (const float*)d_in[26];
    const float* gnn_qb = (const float*)d_in[27];
    const float* gnn_kw = (const float*)d_in[28];
    const float* gnn_kb = (const float*)d_in[29];
    const float* gnn_vw = (const float*)d_in[30];
    const float* gnn_vb = (const float*)d_in[31];
    const float* la_w   = (const float*)d_in[32];
    const float* la_b   = (const float*)d_in[33];
    const float* adj_data = (const float*)d_in[34];
    const int*   adj_row  = (const int*)d_in[35];
    const int*   adj_col  = (const int*)d_in[36];

    // ---- workspace bump allocator (256B aligned) ----
    char* wsb = (char*)d_ws;
    size_t off = 0;
    auto alloc = [&](size_t bytes) -> void* {
        void* p = wsb + off;
        off = (off + bytes + 255) & ~(size_t)255;
        return p;
    };
    // f32 buffers
    float* mm_item_f = (float*)alloc((size_t)NITEMS * DIM * 4);
    float* gl_item   = (float*)alloc((size_t)NITEMS * DIM * 4);
    float* mm_user   = (float*)alloc((size_t)NUSERS * DIM * 4);
    float* gl_user   = (float*)alloc((size_t)NUSERS * DIM * 4);
    float* emb0      = (float*)alloc((size_t)NTOT * DIM * 4);
    float* emb1      = (float*)alloc((size_t)NTOT * DIM * 4);
    float* emb2      = (float*)alloc((size_t)NTOT * DIM * 4);
    float* side      = (float*)alloc((size_t)NTOT * DIM * 4);
    float* vn        = (float*)alloc((size_t)NTOT * DIM * 4);
    float* topw      = (float*)alloc((size_t)NTOT * TOPK * 4);
    int*   topi      = (int*)alloc((size_t)NTOT * TOPK * 4);
    float* finalb    = (float*)alloc((size_t)NTOT * DIM * 4);
    float* relmean   = (float*)alloc(DIM * 4);
    // bf16 activation staging
    bf16_t* mfeat_bf   = (bf16_t*)alloc((size_t)NITEMS * 768 * 2);
    bf16_t* ufeat_bf   = (bf16_t*)alloc((size_t)NUSERS * 512 * 2);
    bf16_t* t_item_bf  = (bf16_t*)alloc((size_t)NITEMS * DIM * 2);
    bf16_t* mm_ie_bf   = (bf16_t*)alloc((size_t)NITEMS * DIM * 2);
    bf16_t* vproj_bf   = (bf16_t*)alloc((size_t)NITEMS * DIM * 2);
    bf16_t* mm_if_bf   = (bf16_t*)alloc((size_t)NITEMS * DIM * 2);
    bf16_t* t_user_bf  = (bf16_t*)alloc((size_t)NUSERS * DIM * 2);
    bf16_t* mm_u_bf    = (bf16_t*)alloc((size_t)NUSERS * DIM * 2);
    bf16_t* item_w_bf  = (bf16_t*)alloc((size_t)NITEMS * DIM * 2);
    bf16_t* user_w_bf  = (bf16_t*)alloc((size_t)NUSERS * DIM * 2);
    bf16_t* emb0_bf    = (bf16_t*)alloc((size_t)NTOT * DIM * 2);
    bf16_t* emb1_bf    = (bf16_t*)alloc((size_t)NTOT * DIM * 2);
    bf16_t* emb2_bf    = (bf16_t*)alloc((size_t)NTOT * DIM * 2);
    bf16_t* side_bf    = (bf16_t*)alloc((size_t)NTOT * DIM * 2);
    bf16_t* sidesc_bf  = (bf16_t*)alloc((size_t)NTOT * DIM * 2);
    bf16_t* qn_bf      = (bf16_t*)alloc((size_t)NTOT * DIM * 2);
    bf16_t* kn_bf      = (bf16_t*)alloc((size_t)NTOT * DIM * 2);
    // bf16 transposed weights
    bf16_t* wt_me1 = (bf16_t*)alloc((size_t)768 * DIM * 2);
    bf16_t* wt_me2 = (bf16_t*)alloc((size_t)128 * DIM * 2);
    bf16_t* wt_cv  = (bf16_t*)alloc((size_t)128 * DIM * 2);
    bf16_t* wt_co  = (bf16_t*)alloc((size_t)128 * DIM * 2);
    bf16_t* wt_ue1 = (bf16_t*)alloc((size_t)512 * DIM * 2);
    bf16_t* wt_ue2 = (bf16_t*)alloc((size_t)128 * DIM * 2);
    bf16_t* wt_mg  = (bf16_t*)alloc((size_t)256 * DIM * 2);
    bf16_t* wt_ug  = (bf16_t*)alloc((size_t)256 * DIM * 2);
    bf16_t* wt_q   = (bf16_t*)alloc((size_t)2 * 128 * DIM * 2);
    bf16_t* wt_k   = (bf16_t*)alloc((size_t)2 * 128 * DIM * 2);
    bf16_t* wt_v   = (bf16_t*)alloc((size_t)2 * 128 * DIM * 2);
    bf16_t* wt_la  = (bf16_t*)alloc((size_t)384 * DIM * 2);
    (void)ws_size; (void)n_in; (void)in_sizes; (void)out_size;

    auto cvt = [&](const float* in, bf16_t* out, int n) {
        k_tobf16<<<(n + 255) / 256, 256, 0, stream>>>(in, out, n);
    };
    auto wtr = [&](const float* w, bf16_t* wt, int K) {
        k_wtrans<<<(K * DIM + 255) / 256, 256, 0, stream>>>(w, wt, K);
    };

    // ---- one-time staging: weights (transpose+bf16) and raw activations ----
    wtr(me_w1, wt_me1, 768);  wtr(me_w2, wt_me2, 128);
    wtr(cm_vw, wt_cv, 128);   wtr(cm_ow, wt_co, 128);
    wtr(ue_w1, wt_ue1, 512);  wtr(ue_w2, wt_ue2, 128);
    wtr(mg_w,  wt_mg, 256);   wtr(ug_w,  wt_ug, 256);
    wtr(gnn_qw, wt_q, 128);   wtr(gnn_qw + 128 * DIM, wt_q + (size_t)128 * DIM, 128);
    wtr(gnn_kw, wt_k, 128);   wtr(gnn_kw + 128 * DIM, wt_k + (size_t)128 * DIM, 128);
    wtr(gnn_vw, wt_v, 128);   wtr(gnn_vw + 128 * DIM, wt_v + (size_t)128 * DIM, 128);
    wtr(la_w, wt_la, 384);
    cvt(mfeat, mfeat_bf, NITEMS * 768);
    cvt(ufeat, ufeat_bf, NUSERS * 512);
    cvt(item_w, item_w_bf, NITEMS * DIM);
    cvt(user_w, user_w_bf, NUSERS * DIM);

#define GEMM(FL, KK, LDWV, Abf, Wt, BIAS, C, CB, M)                             \
    k_gemm128<(FL), (KK), (LDWV)>                                               \
        <<<dim3(((M) + 15) / 16, DIM / 16), 32, 0, stream>>>(                   \
            (Abf), (Wt), (BIAS), (C), (CB), (M))

    // ---- modality encoders ----
    GEMM(F_BIAS | F_RELU | F_OB16, 768, 768, mfeat_bf, wt_me1, me_b1,
         nullptr, t_item_bf, NITEMS);
    GEMM(F_BIAS | F_RELU | F_OB16, 128, 128, t_item_bf, wt_me2, me_b2,
         nullptr, mm_ie_bf, NITEMS);
    // cross-modal attention with duplicated modality reduces exactly to
    // mm_item = 2*((mm_item_e @ cm_vw + cm_vb) @ cm_ow + cm_ob)
    GEMM(F_BIAS | F_OB16, 128, 128, mm_ie_bf, wt_cv, cm_vb,
         nullptr, vproj_bf, NITEMS);
    GEMM(F_BIAS | F_DOUBLE | F_OF32 | F_OB16, 128, 128, vproj_bf, wt_co, cm_ob,
         mm_item_f, mm_if_bf, NITEMS);
    GEMM(F_BIAS | F_RELU | F_OB16, 512, 512, ufeat_bf, wt_ue1, ue_b1,
         nullptr, t_user_bf, NUSERS);
    GEMM(F_BIAS | F_RELU | F_OF32 | F_OB16, 128, 128, t_user_bf, wt_ue2, ue_b2,
         mm_user, mm_u_bf, NUSERS);

    // ---- gated fusion (concat GEMM = two accumulating GEMMs) ----
    GEMM(F_BIAS | F_OF32, 128, 256, item_w_bf, wt_mg, mg_b, gl_item, nullptr, NITEMS);
    GEMM(F_ACC  | F_OF32, 128, 256, mm_if_bf, wt_mg + 128, nullptr, gl_item, nullptr, NITEMS);
    GEMM(F_BIAS | F_OF32, 128, 256, user_w_bf, wt_ug, ug_b, gl_user, nullptr, NUSERS);
    GEMM(F_ACC  | F_OF32, 128, 256, mm_u_bf,  wt_ug + 128, nullptr, gl_user, nullptr, NUSERS);

    {
        const int nu = NUSERS * DIM, ni = NITEMS * DIM;
        k_fuse<<<(nu + 255) / 256, 256, 0, stream>>>(user_w, mm_user, gl_user,
                                                     emb0, emb0_bf, nu);
        k_fuse<<<(ni + 255) / 256, 256, 0, stream>>>(item_w, mm_item_f, gl_item,
                                                     emb0 + (size_t)NUSERS * DIM,
                                                     emb0_bf + (size_t)NUSERS * DIM, ni);
        hipMemcpyAsync(emb0 + (size_t)(NUSERS + NITEMS) * DIM, ent_w,
                       (size_t)NENT * DIM * 4, hipMemcpyDeviceToDevice, stream);
        cvt(ent_w, emb0_bf + (size_t)(NUSERS + NITEMS) * DIM, NENT * DIM);
    }

    k_relmean<<<1, 128, 0, stream>>>(rel_w, relmean);

    // ---- GNN layers ----
    float*  embs  [3] = {emb0, emb1, emb2};
    bf16_t* embsbf[3] = {emb0_bf, emb1_bf, emb2_bf};
    const int nElem = NTOT * DIM;
    for (int l = 0; l < 2; ++l) {
        hipMemsetAsync(side, 0, (size_t)nElem * 4, stream);
        k_segsum<<<NEDGE, 128, 0, stream>>>(adj_data, adj_row, adj_col, embs[l], side);
        cvt(side, side_bf, nElem);
        k_scalecols_bf<<<(nElem + 255) / 256, 256, 0, stream>>>(side, relmean,
                                                                sidesc_bf, nElem);

        const size_t wo = (size_t)l * 128 * DIM;
        GEMM(F_BIAS | F_OB16, 128, 128, embsbf[l], wt_q + wo, gnn_qb + l * DIM,
             nullptr, qn_bf, NTOT);
        GEMM(F_BIAS | F_OB16, 128, 128, sidesc_bf, wt_k + wo, gnn_kb + l * DIM,
             nullptr, kn_bf, NTOT);
        GEMM(F_BIAS | F_OF32, 128, 128, side_bf,   wt_v + wo, gnn_vb + l * DIM,
             vn, nullptr, NTOT);

        k_scores_topk<<<NTOT / 16, 32, 0, stream>>>(qn_bf, kn_bf, topw, topi, NTOT);
        k_agg_relu<<<NTOT, 128, 0, stream>>>(vn, topw, topi, embs[l + 1], embsbf[l + 1]);
    }

    // ---- layer aggregation: concat([e0,e1,e2]) @ la_w + la_b ----
    GEMM(F_BIAS | F_OF32, 128, 384, emb0_bf, wt_la,       la_b,   finalb, nullptr, NTOT);
    GEMM(F_ACC  | F_OF32, 128, 384, emb1_bf, wt_la + 128, nullptr, finalb, nullptr, NTOT);
    GEMM(F_ACC  | F_OF32, 128, 384, emb2_bf, wt_la + 256, nullptr, finalb, nullptr, NTOT);

    // ---- outputs: norm(users), norm(items), item_w, mm_item ----
    float* out = (float*)d_out;
    k_norm<<<NUSERS + NITEMS, 128, 0, stream>>>(finalb, out);
    hipMemcpyAsync(out + (size_t)(NUSERS + NITEMS) * DIM, item_w,
                   (size_t)NITEMS * DIM * 4, hipMemcpyDeviceToDevice, stream);
    hipMemcpyAsync(out + (size_t)(NUSERS + 2 * NITEMS) * DIM, mm_item_f,
                   (size_t)NITEMS * DIM * 4, hipMemcpyDeviceToDevice, stream);
#undef GEMM
}
